// PerceiverAttentionCA_52218212385352
// MI455X (gfx1250) — matrix operand order repaired and checked by reading, not compiled, not run on hardware
//
#include <hip/hip_runtime.h>

typedef __attribute__((ext_vector_type(16))) _Float16     v16h;
typedef __attribute__((ext_vector_type(8)))  float        v8f;
typedef __attribute__((ext_vector_type(4)))  unsigned int v4u;
typedef __attribute__((ext_vector_type(8)))  unsigned int v8u;

union Frag16 {
    v16h v;
    uint4 q[2];
    _Float16 h[16];
};

#define DIM 3072
#define DHID 3072
#define HEADS 48
#define DH 64

// ---------------------------------------------------------------------------
// TDM: 2D tile load Global -> LDS via Tensor Data Mover (gfx1250).
// D# built per CDNA5 ISA. data_size=2 bytes.
// Padding: after every 16 DWORDs (one 32-half row) insert 4 DWORDs, i.e.
// LDS row stride = 40 halves (80B) — matches the fragment addressing below.
// ---------------------------------------------------------------------------
__device__ __forceinline__ void tdm_load_2d_f16(unsigned lds_off,
                                                const void* gptr,
                                                unsigned tensor_d0,
                                                unsigned tensor_d1,
                                                unsigned tile_d0,
                                                unsigned tile_d1,
                                                unsigned stride0) {
    unsigned long long ga = (unsigned long long)(size_t)gptr;
    v4u g0;
    g0[0] = 1u;                                    // count=1, user descriptor
    g0[1] = lds_off;                               // lds_addr (bytes)
    g0[2] = (unsigned)ga;                          // global_addr[31:0]
    g0[3] = (unsigned)(ga >> 32) | (2u << 30);     // global_addr[56:32] | type=2
    v8u g1;
    g1[0] = (1u << 16)        // data_size = 2 bytes
          | (1u << 20)        // pad_enable
          | (3u << 22)        // pad_interval: 16 DWORDs (64B)
          | (3u << 25);       // pad_amount:   4 DWORDs (16B)
    g1[1] = (tensor_d0 & 0xFFFFu) << 16;                       // tensor_dim0 lo
    g1[2] = (tensor_d0 >> 16) | ((tensor_d1 & 0xFFFFu) << 16); // dim0 hi | dim1 lo
    g1[3] = (tensor_d1 >> 16) | (tile_d0 << 16);               // dim1 hi | tile_dim0
    g1[4] = tile_d1;                                           // tile_dim1 (tile_dim2=0)
    g1[5] = stride0;                                           // tensor_dim0_stride lo
    g1[6] = 0u;
    g1[7] = 0u;
    asm volatile("tensor_load_to_lds %0, %1" ::"s"(g0), "s"(g1) : "memory");
}

// ---------------------------------------------------------------------------
// LayerNorm: one block per row of 3072, 256 threads, f32 in -> f16 out
// ---------------------------------------------------------------------------
__global__ __launch_bounds__(256) void ln_kernel(const float* __restrict__ in,
                                                 const float* __restrict__ g,
                                                 const float* __restrict__ b,
                                                 _Float16* __restrict__ out) {
    const int row = blockIdx.x;
    const int t = threadIdx.x;
    const float* r = in + (size_t)row * DIM;
    float vals[12];
    float s = 0.f, ss = 0.f;
#pragma unroll
    for (int i = 0; i < 12; ++i) {
        float v = r[t + i * 256];
        vals[i] = v;
        s += v;
        ss += v * v;
    }
    __shared__ float red1[256];
    __shared__ float red2[256];
    red1[t] = s; red2[t] = ss;
    __syncthreads();
    for (int off = 128; off > 0; off >>= 1) {
        if (t < off) { red1[t] += red1[t + off]; red2[t] += red2[t + off]; }
        __syncthreads();
    }
    const float mu  = red1[0] * (1.f / DIM);
    const float var = red2[0] * (1.f / DIM) - mu * mu;
    const float inv = rsqrtf(var + 1e-5f);
    _Float16* o = out + (size_t)row * DIM;
#pragma unroll
    for (int i = 0; i < 12; ++i) {
        int c = t + i * 256;
        o[c] = (_Float16)((vals[i] - mu) * inv * g[c] + b[c]);
    }
}

// ---------------------------------------------------------------------------
// One-time weight transpose + convert: W[K,N] f32 -> Wt[N,K] f16.
// 64x64 tiles, 256 threads.
// ---------------------------------------------------------------------------
__global__ __launch_bounds__(256) void transpose_cvt(const float* __restrict__ W,
                                                     _Float16* __restrict__ Wt,
                                                     int K, int N) {
    __shared__ _Float16 tile[64][72];   // [k_local][n_local], padded
    const int k0 = blockIdx.x * 64;
    const int n0 = blockIdx.y * 64;
    const int t = threadIdx.x;
    const int r = t >> 2;          // 0..63
    const int c = (t & 3) * 16;    // 0,16,32,48

    const float* src = W + (size_t)(k0 + r) * N + n0 + c;
#pragma unroll
    for (int i = 0; i < 16; i += 4) {
        float4 f = *(const float4*)(src + i);
        tile[r][c + i + 0] = (_Float16)f.x;
        tile[r][c + i + 1] = (_Float16)f.y;
        tile[r][c + i + 2] = (_Float16)f.z;
        tile[r][c + i + 3] = (_Float16)f.w;
    }
    __syncthreads();
    _Float16 buf[16];
#pragma unroll
    for (int i = 0; i < 16; ++i) buf[i] = tile[c + i][r];
    _Float16* dst = Wt + (size_t)(n0 + r) * K + k0 + c;
    *(uint4*)(dst)     = *(uint4*)(&buf[0]);
    *(uint4*)(dst + 8) = *(uint4*)(&buf[8]);
}

// ---------------------------------------------------------------------------
// WMMA GEMM: C[M,N] = A[M,K](f16 rm) * Bt[N,K](f16 rm, pre-transposed weights)
// Block tile 128x128, 8 waves as 4x2, per-wave 32x64 (2x4 WMMA tiles),
// K-step 32. Tiles staged by the Tensor Data Mover into double-buffered LDS;
// wave 0 issues tile i+1 while all waves compute on tile i (DMA/compute
// overlap), then waits TENSORcnt and a barrier releases the block.
// ---------------------------------------------------------------------------
template <bool OUT_F32>
__global__ __launch_bounds__(256) void gemm_f16(const _Float16* __restrict__ A,
                                                const _Float16* __restrict__ Bt,
                                                void* __restrict__ Cout,
                                                int M, int N, int K) {
    constexpr int LS = 40;                 // LDS row stride in halves
    constexpr int ATILE = 128 * LS;        // halves per A buffer
    constexpr int BTILE = 128 * LS;        // halves per B buffer
    __shared__ _Float16 As[2 * ATILE];
    __shared__ _Float16 Bs[2 * BTILE];

    const int t = threadIdx.x;
    const int m0 = blockIdx.x * 128;
    const int n0 = blockIdx.y * 128;
    const int w = t >> 5, l = t & 31;
    const int wm = w & 3;        // 0..3 -> 32-row strip
    const int wn = w >> 2;       // 0..1 -> 64-col strip
    const int laneM = l & 15;
    const int kg = l >> 4;

    const unsigned ldsA = (unsigned)(size_t)&As[0];
    const unsigned ldsB = (unsigned)(size_t)&Bs[0];
    const _Float16* Ab = A + (size_t)m0 * K;
    const _Float16* Bb = Bt + (size_t)n0 * K;

    v8f acc[2][4] = {};

    const int nk = K / 32;
    // prologue: stage tile 0 into buffer 0
    if (w == 0) {
        tdm_load_2d_f16(ldsA, Ab, (unsigned)K, (unsigned)M, 32u, 128u, (unsigned)K);
        tdm_load_2d_f16(ldsB, Bb, (unsigned)K, (unsigned)N, 32u, 128u, (unsigned)K);
        __builtin_amdgcn_s_wait_tensorcnt(0);
    }
    __syncthreads();

    for (int i = 0; i < nk; ++i) {
        const int cur = i & 1;
        // issue DMA for tile i+1 into the other buffer (its last readers were
        // released by the barrier that ended iteration i-1)
        if (w == 0 && i + 1 < nk) {
            int k0 = (i + 1) * 32;
            tdm_load_2d_f16(ldsA + (unsigned)((cur ^ 1) * ATILE * 2), Ab + k0,
                            (unsigned)K, (unsigned)M, 32u, 128u, (unsigned)K);
            tdm_load_2d_f16(ldsB + (unsigned)((cur ^ 1) * BTILE * 2), Bb + k0,
                            (unsigned)K, (unsigned)N, 32u, 128u, (unsigned)K);
        }

        const _Float16* Ap = &As[cur * ATILE];
        const _Float16* Bp = &Bs[cur * BTILE];
        Frag16 a[2], bfr[4];
#pragma unroll
        for (int mt = 0; mt < 2; ++mt) {
            int row = wm * 32 + mt * 16 + laneM;
            a[mt].q[0] = *(const uint4*)(&Ap[row * LS + kg * 8]);
            a[mt].q[1] = *(const uint4*)(&Ap[row * LS + 16 + kg * 8]);
        }
#pragma unroll
        for (int nt = 0; nt < 4; ++nt) {
            int n = wn * 64 + nt * 16 + laneM;
            bfr[nt].q[0] = *(const uint4*)(&Bp[n * LS + kg * 8]);
            bfr[nt].q[1] = *(const uint4*)(&Bp[n * LS + 16 + kg * 8]);
        }
#pragma unroll
        for (int mt = 0; mt < 2; ++mt)
#pragma unroll
            for (int nt = 0; nt < 4; ++nt)
                acc[mt][nt] = __builtin_amdgcn_wmma_f32_16x16x32_f16(
                    false, a[mt].v, false, bfr[nt].v, (short)0, acc[mt][nt],
                    false, false);

        __syncthreads();   // all waves done reading buffer `cur`
        if (w == 0) __builtin_amdgcn_s_wait_tensorcnt(0);
        __syncthreads();   // next tile resident
    }

    // ---- epilogue: C layout M = kg*8+r, N = laneM ----
#pragma unroll
    for (int mt = 0; mt < 2; ++mt)
#pragma unroll
        for (int nt = 0; nt < 4; ++nt)
#pragma unroll
            for (int r = 0; r < 8; ++r) {
                int gr = m0 + wm * 32 + mt * 16 + kg * 8 + r;
                int gc = n0 + wn * 64 + nt * 16 + laneM;
                if (OUT_F32)
                    ((float*)Cout)[(size_t)gr * N + gc] = acc[mt][nt][r];
                else
                    ((_Float16*)Cout)[(size_t)gr * N + gc] =
                        (_Float16)acc[mt][nt][r];
            }
}

// ---------------------------------------------------------------------------
// Attention: one block per (bt, head, 32-row q chunk).
// S = (Q K^T)/8 via WMMA (frags straight from global), softmax in LDS,
// O = P @ V via WMMA with V transposed in LDS.
// ---------------------------------------------------------------------------
__global__ __launch_bounds__(256) void attn_kernel(const _Float16* __restrict__ q,
                                                   const _Float16* __restrict__ kv,
                                                   _Float16* __restrict__ attn) {
    __shared__ _Float16 Vt[DH * 128];   // [d][key], 16 KB
    __shared__ float    Ssc[32 * 128];  // scores, 16 KB
    __shared__ _Float16 P[32 * 128];    // probs,   8 KB

    const int qc = blockIdx.x;   // 0..31
    const int h  = blockIdx.y;   // 0..47
    const int bt = blockIdx.z;   // 0..3
    const int t = threadIdx.x;
    const int w = t >> 5, l = t & 31;
    const int laneM = l & 15;
    const int kg = l >> 4;

    // ---- stage V transposed: Vt[d][key] ----
    {
        int kk = t >> 1;
        int coff = (t & 1) * 32;
        const _Float16* src =
            kv + (size_t)(bt * 128 + kk) * (2 * DHID) + DHID + h * DH + coff;
#pragma unroll
        for (int c = 0; c < 32; ++c) Vt[(coff + c) * 128 + kk] = src[c];
    }

    // ---- S = Q K^T : M=32 (2 tiles), N=128 (8 tiles), K=64 (2 steps) ----
    const int wm = w & 1;
    const int wn = w >> 1;
    v8f accS[2] = {};
#pragma unroll
    for (int ks = 0; ks < 2; ++ks) {
        Frag16 a;
        {
            size_t qrow = (size_t)(bt * 1024 + qc * 32 + wm * 16 + laneM);
            const _Float16* qp = q + qrow * DHID + h * DH + ks * 32;
            a.q[0] = *(const uint4*)(qp + kg * 8);
            a.q[1] = *(const uint4*)(qp + 16 + kg * 8);
        }
#pragma unroll
        for (int j = 0; j < 2; ++j) {
            int nt = wn * 2 + j;
            Frag16 bf;
            {
                int key = nt * 16 + laneM;
                const _Float16* kp =
                    kv + (size_t)(bt * 128 + key) * (2 * DHID) + h * DH +
                    ks * 32 + kg * 8;
                bf.q[0] = *(const uint4*)(kp);
                bf.q[1] = *(const uint4*)(kp + 16);
            }
            accS[j] = __builtin_amdgcn_wmma_f32_16x16x32_f16(
                false, a.v, false, bf.v, (short)0, accS[j], false, false);
        }
    }
#pragma unroll
    for (int j = 0; j < 2; ++j) {
        int nt = wn * 2 + j;
#pragma unroll
        for (int r = 0; r < 8; ++r) {
            int srow = wm * 16 + kg * 8 + r;
            int scol = nt * 16 + laneM;
            Ssc[srow * 128 + scol] = accS[j][r] * 0.125f;
        }
    }
    __syncthreads();

    // ---- softmax over 128 keys, one thread per row ----
    if (t < 32) {
        float m = -3.0e38f;
        for (int c = 0; c < 128; ++c) m = fmaxf(m, Ssc[t * 128 + c]);
        float s = 0.f;
        for (int c = 0; c < 128; ++c) {
            float e = __expf(Ssc[t * 128 + c] - m);
            Ssc[t * 128 + c] = e;
            s += e;
        }
        float inv = 1.f / s;
        for (int c = 0; c < 128; ++c)
            P[t * 128 + c] = (_Float16)(Ssc[t * 128 + c] * inv);
    }
    __syncthreads();

    // ---- O = P @ V : M=32 (2 tiles), N=64 (4 tiles), K=128 (4 steps) ----
    const int wm2 = w & 1;
    const int wn2 = w >> 1;
    v8f accO = {};
#pragma unroll
    for (int ks = 0; ks < 4; ++ks) {
        Frag16 a, bf;
        {
            int row = wm2 * 16 + laneM;
            a.q[0] = *(const uint4*)(&P[row * 128 + ks * 32 + kg * 8]);
            a.q[1] = *(const uint4*)(&P[row * 128 + ks * 32 + 16 + kg * 8]);
        }
        {
            int n = wn2 * 16 + laneM;
            bf.q[0] = *(const uint4*)(&Vt[n * 128 + ks * 32 + kg * 8]);
            bf.q[1] = *(const uint4*)(&Vt[n * 128 + ks * 32 + 16 + kg * 8]);
        }
        accO = __builtin_amdgcn_wmma_f32_16x16x32_f16(
            false, a.v, false, bf.v, (short)0, accO, false, false);
    }
#pragma unroll
    for (int r = 0; r < 8; ++r) {
        size_t gr = (size_t)(bt * 1024 + qc * 32 + wm2 * 16 + kg * 8 + r);
        int gc = h * DH + wn2 * 16 + laneM;
        attn[gr * DHID + gc] = (_Float16)accO[r];
    }
}

// ---------------------------------------------------------------------------
extern "C" void kernel_launch(void* const* d_in, const int* in_sizes, int n_in,
                              void* d_out, int out_size, void* d_ws, size_t ws_size,
                              hipStream_t stream) {
    const float* x       = (const float*)d_in[0];  // [4,128,3072]
    const float* latents = (const float*)d_in[1];  // [4,1024,3072]
    const float* g1 = (const float*)d_in[2];
    const float* b1 = (const float*)d_in[3];
    const float* g2 = (const float*)d_in[4];
    const float* b2 = (const float*)d_in[5];
    const float* Wq  = (const float*)d_in[6];  // [3072,3072]
    const float* Wkv = (const float*)d_in[7];  // [3072,6144]
    const float* Wo  = (const float*)d_in[8];  // [3072,3072]
    float* out = (float*)d_out;                // [4,1024,3072]

    size_t off = 0;
    auto carve = [&](size_t bytes) -> char* {
        char* p = (char*)d_ws + off;
        off += (bytes + 255) & ~(size_t)255;
        return p;
    };
    _Float16* ln_lat = (_Float16*)carve((size_t)4096 * DIM * 2);      // 24 MB
    _Float16* xn     = (_Float16*)carve((size_t)512 * DIM * 2);       //  3 MB
    _Float16* qbuf   = (_Float16*)carve((size_t)4096 * DHID * 2);     // 24 MB
    _Float16* kvbuf  = (_Float16*)carve((size_t)512 * 2 * DHID * 2);  //  6 MB
    _Float16* WqT    = (_Float16*)carve((size_t)DHID * DIM * 2);      // 18 MB
    _Float16* WkvT   = (_Float16*)carve((size_t)2 * DHID * DIM * 2);  // 36 MB
    _Float16* WoT    = (_Float16*)carve((size_t)DIM * DHID * 2);      // 18 MB
    _Float16* attn   = ln_lat;  // ln_lat is dead after the q projection

    // 0) one-time weight transpose+convert: W[K,N] f32 -> Wt[N,K] f16
    transpose_cvt<<<dim3(DIM / 64, DHID / 64), 256, 0, stream>>>(Wq, WqT, DIM, DHID);
    transpose_cvt<<<dim3(DIM / 64, 2 * DHID / 64), 256, 0, stream>>>(Wkv, WkvT, DIM, 2 * DHID);
    transpose_cvt<<<dim3(DHID / 64, DIM / 64), 256, 0, stream>>>(Wo, WoT, DHID, DIM);

    // 1) LayerNorms
    ln_kernel<<<512, 256, 0, stream>>>(x, g1, b1, xn);
    ln_kernel<<<4096, 256, 0, stream>>>(latents, g2, b2, ln_lat);

    // 2) kv = xn @ Wkv : [512,3072] x [3072,6144] -> f16
    gemm_f16<false><<<dim3(512 / 128, 6144 / 128), 256, 0, stream>>>(
        xn, WkvT, kvbuf, 512, 6144, DIM);

    // 3) q = ln @ Wq : [4096,3072] x [3072,3072] -> f16
    gemm_f16<false><<<dim3(4096 / 128, 3072 / 128), 256, 0, stream>>>(
        ln_lat, WqT, qbuf, 4096, 3072, DIM);

    // 4) attention per (bt, head, 32-row q chunk)
    attn_kernel<<<dim3(32, HEADS, 4), 256, 0, stream>>>(qbuf, kvbuf, attn);

    // 5) out = attn @ Wo : [4096,3072] x [3072,3072] -> f32 output
    gemm_f16<true><<<dim3(4096 / 128, 3072 / 128), 256, 0, stream>>>(
        attn, WoT, out, 4096, 3072, DIM);
}
